// LinearComponent_60559038874131
// MI455X (gfx1250) — compile-verified, matrix operand order checked by reading, not run on hardware
//
#include <hip/hip_runtime.h>
#include <stdint.h>

// ---------------------------------------------------------------------------
// LinearComponent: out[b,i,o] = sum_c mask[b,i,c] * (x[b,i,:] @ A[i,c]) @ B[i,c]
// Two bf16-WMMA GEMM passes through a 64MB bf16 workspace.
//   Pass 1: inner[i][b][c*256+m] = mask * (X_i @ A[i,c])      (bf16 store)
//   Pass 2: out_i = inner_i[512x16384] @ Bflat_i[16384x1024]  (component sum
//           fused into K; B[i] is already [16384,1024] row-major)
// f32 operands converted to bf16 on the fly; f32 accumulation in WMMA.
// Pass-2 A tiles stream via global_load_async_to_lds_b128 (ASYNCcnt).
// ---------------------------------------------------------------------------

#define B_SZ   512
#define N_INST 4
#define D_IN   1024
#define D_OUT  1024
#define C_COMP 64
#define M_RANK 256
#define K_FUSED (C_COMP * M_RANK)   // 16384

typedef __attribute__((ext_vector_type(16))) __bf16 v16bf;
typedef __attribute__((ext_vector_type(8)))  float  v8f;

__device__ __forceinline__ uint32_t pack2bf(float a, float b) {
    union { __bf16 h[2]; uint32_t u; } p;
    p.h[0] = (__bf16)a; p.h[1] = (__bf16)b;
    return p.u;
}

__device__ __forceinline__ v8f wmma_bf16(v16bf a, v16bf b, v8f c) {
    // D = A(16x32 bf16) * B(32x16 bf16) + C(16x16 f32)
    return __builtin_amdgcn_wmma_f32_16x16x32_bf16(
        /*neg_a=*/false, a, /*neg_b=*/false, b,
        /*c_mod=*/(short)0, c, /*reuse_a=*/false, /*reuse_b=*/false);
}

// Fragment gather from an LDS tile stored [row][k], pitch P bf16 elements
// (P % 8 == 0 so each 16B chunk is aligned -> 2x ds_load_b128 per fragment).
// Per ISA 7.12.2 (16-bit A/B operand): lanes 0-15 hold row (lane&15) with
// K = {0..7, 16..23}; lanes 16-31 hold the same rows with K = {8..15, 24..31}.
template<int P>
__device__ __forceinline__ v16bf load_frag(const __bf16* s, int rowBase) {
    static_assert(P % 8 == 0, "pitch must keep 16B alignment");
    const int lane = threadIdx.x & 31;
    const int half = lane >> 4;
    const int r    = lane & 15;
    const __bf16* p = s + (rowBase + r) * P + half * 8;
    union { v16bf v; uint4 q[2]; } f;
    f.q[0] = *(const uint4*)p;
    f.q[1] = *(const uint4*)(p + 16);
    return f.v;
}

__device__ __forceinline__ void async_load_lds_b128(uint32_t lds_off, const void* gaddr) {
    // CDNA5 async global->LDS copy (tracked by ASYNCcnt). One 16B line/lane.
    asm volatile("global_load_async_to_lds_b128 %0, %1, off"
                 :: "v"(lds_off), "v"(gaddr) : "memory");
}

__device__ __forceinline__ void wait_asynccnt0() {
    asm volatile("s_wait_asynccnt 0x0" ::: "memory");
}

// ---------------------------------------------------------------------------
// Pass 1: inner[i][b][c*256+m] = mask[b,i,c] * (x[b,i,:] @ A[i,c])   (bf16)
// GEMM M=512(b) K=1024(d) N=256(m) per (i,c). BM=128 BN=128 BK=32.
// grid: x = 4 mTiles * 2 nTiles = 8, y = c (64), z = i (4). 256 threads.
// ---------------------------------------------------------------------------
__global__ __launch_bounds__(256) void lc_pass1_inner(
    const float* __restrict__ x,      // [512,4,1024]
    const float* __restrict__ A,      // [4,64,1024,256]
    const int*   __restrict__ mask,   // [512,4,64]
    __bf16*      __restrict__ inner)  // [4,512,16384]
{
    constexpr int BM = 128, BN = 128, BK = 32, P = 40, KT = D_IN / BK;
    const int bTile = blockIdx.x & 3;
    const int nTile = blockIdx.x >> 2;
    const int c     = blockIdx.y;
    const int inst  = blockIdx.z;
    const int tid   = threadIdx.x;
    const int lane  = tid & 31;
    const int w     = tid >> 5;
    const int wm    = w & 3;    // 4 wave-rows  -> 32 rows each
    const int wn    = w >> 2;   // 2 wave-cols  -> 64 cols each

    __shared__ __attribute__((aligned(16))) __bf16 Xs[2][BM * P]; // [row][k]
    __shared__ __attribute__((aligned(16))) __bf16 Ws[2][BN * P]; // [n][k] (k-transposed)

    const int row0 = bTile * BM;
    const int n0   = nTile * BN;
    const float* Abase = A + ((size_t)(inst * C_COMP + c)) * D_IN * M_RANK;

    const int xrow = tid >> 1, xpart = tid & 1;   // X tile: 128 rows x 32 f32
    const int wrow = tid >> 3, wpart = tid & 7;   // A tile: 32 rows x 128 f32

    float4 xr[4], ar[4];
    auto gload = [&](int kt) {
        const int k0 = kt * BK;
        const float* xp = x + ((size_t)(row0 + xrow) * N_INST + inst) * D_IN + k0 + xpart * 16;
        #pragma unroll
        for (int q = 0; q < 4; ++q) xr[q] = ((const float4*)xp)[q];
        const float* ap = Abase + (size_t)(k0 + wrow) * M_RANK + n0 + wpart * 16;
        #pragma unroll
        for (int q = 0; q < 4; ++q) ar[q] = ((const float4*)ap)[q];
    };
    auto sstore = [&](int buf) {
        // X rows: two 16B packed stores per thread.
        uint4* xd = (uint4*)&Xs[buf][xrow * P + xpart * 16];
        uint4 p0, p1;
        p0.x = pack2bf(xr[0].x, xr[0].y); p0.y = pack2bf(xr[0].z, xr[0].w);
        p0.z = pack2bf(xr[1].x, xr[1].y); p0.w = pack2bf(xr[1].z, xr[1].w);
        p1.x = pack2bf(xr[2].x, xr[2].y); p1.y = pack2bf(xr[2].z, xr[2].w);
        p1.z = pack2bf(xr[3].x, xr[3].y); p1.w = pack2bf(xr[3].z, xr[3].w);
        xd[0] = p0; xd[1] = p1;
        // Weights: k-transposed scatter (2B stores).
        #pragma unroll
        for (int q = 0; q < 4; ++q) {
            const int n = wpart * 16 + q * 4;
            Ws[buf][(n + 0) * P + wrow] = (__bf16)ar[q].x;
            Ws[buf][(n + 1) * P + wrow] = (__bf16)ar[q].y;
            Ws[buf][(n + 2) * P + wrow] = (__bf16)ar[q].z;
            Ws[buf][(n + 3) * P + wrow] = (__bf16)ar[q].w;
        }
    };

    v8f acc[2][4];
    const v8f vzero = {0.f, 0.f, 0.f, 0.f, 0.f, 0.f, 0.f, 0.f};
    #pragma unroll
    for (int i = 0; i < 2; ++i)
        #pragma unroll
        for (int j = 0; j < 4; ++j) acc[i][j] = vzero;

    gload(0); sstore(0); __syncthreads();

    for (int kt = 0; kt < KT; ++kt) {
        const int cur = kt & 1;
        if (kt + 1 < KT) gload(kt + 1);

        v16bf afr[2], bfr[4];
        #pragma unroll
        for (int i = 0; i < 2; ++i) afr[i] = load_frag<P>(Xs[cur], wm * 32 + i * 16);
        #pragma unroll
        for (int j = 0; j < 4; ++j) bfr[j] = load_frag<P>(Ws[cur], wn * 64 + j * 16);
        #pragma unroll
        for (int i = 0; i < 2; ++i)
            #pragma unroll
            for (int j = 0; j < 4; ++j)
                acc[i][j] = wmma_bf16(afr[i], bfr[j], acc[i][j]);

        if (kt + 1 < KT) sstore(cur ^ 1);
        __syncthreads();
    }

    // Epilogue: apply per-row component mask, store bf16 into fused-K workspace.
    const int lc = lane & 15, lh = lane >> 4;
    #pragma unroll
    for (int i = 0; i < 2; ++i) {
        #pragma unroll
        for (int r = 0; r < 8; ++r) {
            const int gRow = row0 + wm * 32 + i * 16 + lh * 8 + r;
            const int mk = mask[((size_t)gRow * N_INST + inst) * C_COMP + c];
            __bf16* op = inner + ((size_t)(inst * B_SZ + gRow)) * K_FUSED
                               + c * M_RANK + n0 + wn * 64 + lc;
            #pragma unroll
            for (int j = 0; j < 4; ++j) {
                const float v = mk ? acc[i][j][r] : 0.0f;
                op[j * 16] = (__bf16)v;
            }
        }
    }
}

// ---------------------------------------------------------------------------
// Pass 2: out_i[512,1024] = inner_i[512,16384] @ Bflat_i[16384,1024]
// BM=64 BN=128 BK=32, 512 K-steps. A tiles (bf16) streamed to LDS with
// global_load_async_to_lds_b128; B tiles converted f32->bf16 through VGPRs.
// grid: x = 8 mTiles * 8 nTiles = 64, z = i (4). 256 threads.
// ---------------------------------------------------------------------------
__global__ __launch_bounds__(256) void lc_pass2_out(
    const __bf16* __restrict__ inner, // [4,512,16384]
    const float*  __restrict__ Bw,    // [4,16384,1024]
    float*        __restrict__ out)   // [512,4,1024]
{
    constexpr int BM = 64, BN = 128, BK = 32, PA = 32, PB = 40, KT = K_FUSED / BK;
    const int bTile = blockIdx.x & 7;
    const int nTile = blockIdx.x >> 3;
    const int inst  = blockIdx.z;
    const int tid   = threadIdx.x;
    const int lane  = tid & 31;
    const int w     = tid >> 5;
    const int wm    = w & 1;    // 2 wave-rows -> 32 rows each
    const int wn    = w >> 1;   // 4 wave-cols -> 32 cols each

    __shared__ __attribute__((aligned(16))) __bf16 Xs[2][BM * PA]; // bf16 A tile [row][k]
    __shared__ __attribute__((aligned(16))) __bf16 Ws[2][BN * PB]; // [n][k]

    const int row0 = bTile * BM;
    const int n0   = nTile * BN;
    const __bf16* Ibase = inner + (size_t)(inst * B_SZ) * K_FUSED;
    const float*  Bbase = Bw + (size_t)inst * K_FUSED * D_OUT;

    const int arow = tid >> 2, apart = tid & 3;   // 64 rows x (32 bf16 = 4 x 16B)
    const int brow = tid >> 3, bpart = tid & 7;   // 32 rows x 128 f32

    // Per-thread LDS destinations for the async A-tile copy (byte offsets:
    // flat pointers to LDS carry the LDS offset in their low 32 bits).
    const uint32_t ldsA0 = (uint32_t)(uintptr_t)&Xs[0][arow * PA + apart * 8];
    const uint32_t ldsA1 = (uint32_t)(uintptr_t)&Xs[1][arow * PA + apart * 8];

    auto aload_async = [&](int kt, int buf) {
        const __bf16* gp = Ibase + (size_t)(row0 + arow) * K_FUSED
                                 + (size_t)kt * BK + apart * 8;
        async_load_lds_b128(buf ? ldsA1 : ldsA0, gp);
    };

    float4 br[4];
    auto gloadW = [&](int kt) {
        const float* bp = Bbase + ((size_t)kt * BK + brow) * D_OUT + n0 + bpart * 16;
        #pragma unroll
        for (int q = 0; q < 4; ++q) br[q] = ((const float4*)bp)[q];
        __builtin_prefetch(bp + (size_t)BK * D_OUT, 0, 0);  // global_prefetch_b8
    };
    auto sstoreW = [&](int buf) {
        #pragma unroll
        for (int q = 0; q < 4; ++q) {
            const int n = bpart * 16 + q * 4;
            Ws[buf][(n + 0) * PB + brow] = (__bf16)br[q].x;
            Ws[buf][(n + 1) * PB + brow] = (__bf16)br[q].y;
            Ws[buf][(n + 2) * PB + brow] = (__bf16)br[q].z;
            Ws[buf][(n + 3) * PB + brow] = (__bf16)br[q].w;
        }
    };

    v8f acc[2][2];
    const v8f vzero = {0.f, 0.f, 0.f, 0.f, 0.f, 0.f, 0.f, 0.f};
    #pragma unroll
    for (int i = 0; i < 2; ++i)
        #pragma unroll
        for (int j = 0; j < 2; ++j) acc[i][j] = vzero;

    aload_async(0, 0);
    gloadW(0);
    sstoreW(0);
    wait_asynccnt0();
    __syncthreads();

    for (int kt = 0; kt < KT; ++kt) {
        const int cur = kt & 1;
        if (kt + 1 < KT) {
            aload_async(kt + 1, cur ^ 1);   // async copy overlaps this tile's math
            gloadW(kt + 1);
        }

        v16bf afr[2], bfr[2];
        #pragma unroll
        for (int i = 0; i < 2; ++i) afr[i] = load_frag<PA>(Xs[cur], wm * 32 + i * 16);
        #pragma unroll
        for (int j = 0; j < 2; ++j) bfr[j] = load_frag<PB>(Ws[cur], wn * 32 + j * 16);
        #pragma unroll
        for (int i = 0; i < 2; ++i)
            #pragma unroll
            for (int j = 0; j < 2; ++j)
                acc[i][j] = wmma_bf16(afr[i], bfr[j], acc[i][j]);

        if (kt + 1 < KT) sstoreW(cur ^ 1);
        wait_asynccnt0();                   // next A tile resident before barrier
        __syncthreads();
    }

    const int lc = lane & 15, lh = lane >> 4;
    #pragma unroll
    for (int i = 0; i < 2; ++i) {
        #pragma unroll
        for (int j = 0; j < 2; ++j) {
            const int gCol = n0 + wn * 32 + j * 16 + lc;
            #pragma unroll
            for (int r = 0; r < 8; ++r) {
                const int gRow = row0 + wm * 32 + i * 16 + lh * 8 + r;
                out[((size_t)gRow * N_INST + inst) * D_OUT + gCol] = acc[i][j][r];
            }
        }
    }
}

// ---------------------------------------------------------------------------
extern "C" void kernel_launch(void* const* d_in, const int* in_sizes, int n_in,
                              void* d_out, int out_size, void* d_ws, size_t ws_size,
                              hipStream_t stream) {
    const float* x    = (const float*)d_in[0];
    const float* A    = (const float*)d_in[1];
    const float* Bw   = (const float*)d_in[2];
    const int*   mask = (const int*)d_in[3];
    float* out = (float*)d_out;

    // Workspace: masked inner activations, bf16 [4][512][16384] = 64 MB.
    __bf16* inner = (__bf16*)d_ws;

    dim3 g1(8, C_COMP, N_INST);   // 2048 workgroups
    lc_pass1_inner<<<g1, 256, 0, stream>>>(x, A, mask, inner);

    dim3 g2(64, 1, N_INST);       // 256 workgroups
    lc_pass2_out<<<g2, 256, 0, stream>>>(inner, Bw, out);

    (void)in_sizes; (void)n_in; (void)out_size; (void)ws_size;
}
